// MoETransformer_21981642621063
// MI455X (gfx1250) — matrix operand order repaired and checked by reading, not compiled, not run on hardware
//
#include <hip/hip_runtime.h>
#include <math.h>

// ---------- types ----------
typedef __attribute__((ext_vector_type(16))) __bf16 bf16x16;
typedef __attribute__((ext_vector_type(8)))  float  f32x8;
typedef __attribute__((ext_vector_type(4)))  float  f32x4;

// ---------- constants ----------
#define TOK   4096      // B*S tokens
#define DMODEL 1024
#define SEQ   512
#define NHEAD 16
#define DHEAD 64
#define NEXP  8
#define CAP   1280      // ceil(2*4096/8*1.25)
#define DFF   4096

// LDS tile strides (floats). Chosen so every 16B fragment chunk stays 16B aligned
// (36*4=144, 68*4=272 are multiples of 16) while spreading LDS banks.
#define A_STRIDE 36     // A tile rows: 32 data floats + pad
#define BT_STRIDE 36    // B (k-contig) rows: 32 data floats + pad
#define BR_STRIDE 68    // B (n-contig) rows: 64 data floats + pad

// ---------- generic batched WMMA GEMM ----------
// C[m,n] = sum_k A[m,k] * B(k,n)  (+bias[n]) (+gelu)
// A row-major with lda. B element (k,n) at B[k*bsk + n*bsn].
// BROW template: true -> B row-major (bsn==1), false -> B transposed weight (bsk==1).
// Per grid.z batch: ptr += (z/zdiv)*Outer + (z%zdiv)*Inner.
// Block tile 128(M) x 64(N); 8 wave32s; each wave owns a 32x32 patch -> 4 WMMAs/K-step.
// Pipeline: double-buffered async DMA (global -> LDS) overlapped with WMMA compute.
struct GemmP {
  const float* A; const float* B; const float* bias; float* C;
  int K, lda, bsk, bsn, ldc, zdiv, epi, pad;
  long aO, aI, bO, bI, cO, cI, biasO, biasI;
};

__device__ inline float gelu_tanh(float x) {
  float x3 = x * x * x;
  return 0.5f * x * (1.0f + tanhf(0.7978845608028654f * (x + 0.044715f * x3)));
}

// async copy of one 16-byte chunk: global -> LDS (tracked by ASYNCcnt)
__device__ inline void async_cp16(void* lds_ptr, const float* gptr) {
  unsigned lds = (unsigned)(unsigned long long)lds_ptr;          // low 32 bits = LDS offset
  unsigned long long ga = (unsigned long long)gptr;
  asm volatile("global_load_async_to_lds_b128 %0, %1, off"
               :: "v"(lds), "v"(ga) : "memory");
}

__device__ inline void wait_async0() {
  asm volatile("s_wait_asynccnt 0x0" ::: "memory");
}

// convert two 8-float runs in LDS into one packed bf16x16 fragment
__device__ inline bf16x16 cvt16(const float* a, const float* b) {
  alignas(32) __bf16 t[16];
  #pragma unroll
  for (int i = 0; i < 8; ++i) t[i] = (__bf16)a[i];
  #pragma unroll
  for (int i = 0; i < 8; ++i) t[8 + i] = (__bf16)b[i];
  return *(const bf16x16*)t;
}

// strided gather (B tile stored k-major) -> packed bf16x16 fragment
__device__ inline bf16x16 cvt16_strided(const float* base, int stride) {
  alignas(32) __bf16 t[16];
  #pragma unroll
  for (int i = 0; i < 16; ++i) t[i] = (__bf16)base[i * stride];
  return *(const bf16x16*)t;
}

template <bool BROW>
__device__ inline void stage_async(float* Af, float* Bf,
                                   const float* A, const float* B, const GemmP& p,
                                   int tileM, int tileN, int kt, int tid) {
  // A tile 128x32 fp32: 1024 x 16B chunks, 4 per thread
  #pragma unroll
  for (int c = 0; c < 4; ++c) {
    const int id  = c * 256 + tid;
    const int row = id >> 3;
    const int kc  = (id & 7) * 4;
    async_cp16(&Af[row * A_STRIDE + kc],
               A + (long)(tileM + row) * p.lda + kt + kc);
  }
  // prefetch one tile beyond the in-flight DMA (L2 warm; speculative)
  __builtin_prefetch(A + (long)(tileM + (tid >> 1)) * p.lda + kt + 32, 0, 1);

  // B tile 32(K)x64(N) fp32: 512 x 16B chunks, 2 per thread
  if constexpr (BROW) {          // global layout [k][n-contig] -> LDS [k][BR_STRIDE]
    #pragma unroll
    for (int c = 0; c < 2; ++c) {
      const int id = c * 256 + tid;
      const int k  = id >> 4;
      const int nc = (id & 15) * 4;
      async_cp16(&Bf[k * BR_STRIDE + nc],
                 B + (long)(kt + k) * p.bsk + tileN + nc);
    }
    __builtin_prefetch(B + (long)(kt + 32 + (tid >> 4)) * p.bsk + tileN, 0, 1);
  } else {                       // global layout [n][k-contig] -> LDS [n][BT_STRIDE]
    #pragma unroll
    for (int c = 0; c < 2; ++c) {
      const int id = c * 256 + tid;
      const int n  = id >> 3;
      const int kc = (id & 7) * 4;
      async_cp16(&Bf[n * BT_STRIDE + kc],
                 B + (long)(tileN + n) * p.bsn + kt + kc);
    }
    __builtin_prefetch(B + (long)(tileN + (tid >> 2)) * p.bsn + kt + 32, 0, 1);
  }
}

template <bool BROW>
__global__ __launch_bounds__(256) void gemm_bf16_wmma(GemmP p) {
  const int z = blockIdx.z;
  const long zq = (long)(z / p.zdiv);
  const long zr = (long)(z % p.zdiv);
  const float* A = p.A + zq * p.aO + zr * p.aI;
  const float* B = p.B + zq * p.bO + zr * p.bI;
  float*       C = p.C + zq * p.cO + zr * p.cI;
  const float* bias = p.bias ? (p.bias + zq * p.biasO + zr * p.biasI) : nullptr;

  const int tileN = blockIdx.x * 64;
  const int tileM = blockIdx.y * 128;

  // double-buffered fp32 staging tiles
  __shared__ float Af[2][128 * A_STRIDE];     // 2 x 18KB
  __shared__ float Bf[2][64 * BR_STRIDE / 1 > 64 * BT_STRIDE ? 64 * BR_STRIDE / 1 : 64 * BT_STRIDE]; // max view
  // note: BROW view uses [32][BR_STRIDE] (2176 floats), !BROW uses [64][BT_STRIDE] (2304)

  const int tid  = threadIdx.x;
  const int lane = tid & 31;
  const int wave = tid >> 5;
  const int wr   = (wave & 3) * 32;   // wave row base within tile (0/32/64/96)
  const int wc   = (wave >> 2) * 32;  // wave col base within tile (0/32)
  const int half = lane >> 4;
  const int l16  = lane & 15;

  f32x8 acc00 = {0.f,0.f,0.f,0.f,0.f,0.f,0.f,0.f};
  f32x8 acc01 = {0.f,0.f,0.f,0.f,0.f,0.f,0.f,0.f};
  f32x8 acc10 = {0.f,0.f,0.f,0.f,0.f,0.f,0.f,0.f};
  f32x8 acc11 = {0.f,0.f,0.f,0.f,0.f,0.f,0.f,0.f};

  // prologue: kick off DMA for first K-tile
  stage_async<BROW>(Af[0], Bf[0], A, B, p, tileM, tileN, 0, tid);

  int buf = 0;
  for (int kt = 0; kt < p.K; kt += 32) {
    wait_async0();        // our DMA chunks for `buf` have landed in LDS
    __syncthreads();      // everyone's landed; also: all waves done reading buf^1

    if (kt + 32 < p.K)    // kick off DMA for next K-tile into the other buffer
      stage_async<BROW>(Af[buf ^ 1], Bf[buf ^ 1], A, B, p, tileM, tileN, kt + 32, tid);

    // ---- build fragments (fp32 LDS -> packed bf16) in contiguous k order (k = 16*half + s) ----
    // A 16x32: lanes0-15 hold K0..15 (V0..7), lanes16-31 hold K16..31
    const float* a0r = &Af[buf][(wr + l16) * A_STRIDE];
    const float* a1r = &Af[buf][(wr + 16 + l16) * A_STRIDE];
    bf16x16 a0 = cvt16(a0r + half * 16, a0r + half * 16 + 8);
    bf16x16 a1 = cvt16(a1r + half * 16, a1r + half * 16 + 8);
    // B 32x16: lanes0-15 hold K0..15 (V0..7), lanes16-31 hold K16..31
    bf16x16 b0, b1;
    if constexpr (BROW) {  // LDS [k][n]: gather strided along k
      b0 = cvt16_strided(&Bf[buf][(half * 16) * BR_STRIDE + wc + l16],      BR_STRIDE);
      b1 = cvt16_strided(&Bf[buf][(half * 16) * BR_STRIDE + wc + 16 + l16], BR_STRIDE);
    } else {               // LDS [n][k]: contiguous along k
      const float* b0r = &Bf[buf][(wc + l16) * BT_STRIDE];
      const float* b1r = &Bf[buf][(wc + 16 + l16) * BT_STRIDE];
      b0 = cvt16(b0r + half * 16, b0r + half * 16 + 8);
      b1 = cvt16(b1r + half * 16, b1r + half * 16 + 8);
    }

    acc00 = __builtin_amdgcn_wmma_f32_16x16x32_bf16(false, a0, false, b0,
                                                    (short)0, acc00, false, false);
    acc01 = __builtin_amdgcn_wmma_f32_16x16x32_bf16(false, a0, false, b1,
                                                    (short)0, acc01, false, false);
    acc10 = __builtin_amdgcn_wmma_f32_16x16x32_bf16(false, a1, false, b0,
                                                    (short)0, acc10, false, false);
    acc11 = __builtin_amdgcn_wmma_f32_16x16x32_bf16(false, a1, false, b1,
                                                    (short)0, acc11, false, false);
    buf ^= 1;
  }

  // ---- epilogue: C layout Vr: lanes0-15 -> M=r, lanes16-31 -> M=8+r; N=lane%16
  const int n0 = tileN + wc + l16;
  const int n1 = n0 + 16;
  float bi0 = 0.f, bi1 = 0.f;
  if (bias) { bi0 = bias[n0]; bi1 = bias[n1]; }
  #pragma unroll
  for (int r = 0; r < 8; ++r) {
    const int m0 = tileM + wr + half * 8 + r;
    const int m1 = m0 + 16;
    float v00 = acc00[r] + bi0;
    float v01 = acc01[r] + bi1;
    float v10 = acc10[r] + bi0;
    float v11 = acc11[r] + bi1;
    if (p.epi == 1) {
      v00 = gelu_tanh(v00); v01 = gelu_tanh(v01);
      v10 = gelu_tanh(v10); v11 = gelu_tanh(v11);
    }
    C[(long)m0 * p.ldc + n0] = v00;
    C[(long)m0 * p.ldc + n1] = v01;
    C[(long)m1 * p.ldc + n0] = v10;
    C[(long)m1 * p.ldc + n1] = v11;
  }
}

// ---------- block reductions (256 threads, wave32) ----------
__device__ inline float blockSum(float v) {
  __shared__ float s[8];
  const int lane = threadIdx.x & 31, w = threadIdx.x >> 5;
  #pragma unroll
  for (int o = 16; o > 0; o >>= 1) v += __shfl_down(v, o);
  __syncthreads();
  if (lane == 0) s[w] = v;
  __syncthreads();
  float r = s[0];
  #pragma unroll
  for (int i = 1; i < 8; ++i) r += s[i];
  return r;
}

__device__ inline float blockMax(float v) {
  __shared__ float s[8];
  const int lane = threadIdx.x & 31, w = threadIdx.x >> 5;
  #pragma unroll
  for (int o = 16; o > 0; o >>= 1) v = fmaxf(v, __shfl_down(v, o));
  __syncthreads();
  if (lane == 0) s[w] = v;
  __syncthreads();
  float r = s[0];
  #pragma unroll
  for (int i = 1; i < 8; ++i) r = fmaxf(r, s[i]);
  return r;
}

// ---------- softmax over 512-wide score rows, scaled by 1/sqrt(64) ----------
__global__ __launch_bounds__(256) void softmax512_kernel(float* scores) {
  float* p = scores + (long)blockIdx.x * 512;
  float a = p[threadIdx.x]       * 0.125f;
  float b = p[threadIdx.x + 256] * 0.125f;
  float mx = blockMax(fmaxf(a, b));
  a = expf(a - mx); b = expf(b - mx);
  float inv = 1.0f / blockSum(a + b);
  p[threadIdx.x]       = a * inv;
  p[threadIdx.x + 256] = b * inv;
}

// ---------- out = LayerNorm(x + r) ----------
__global__ __launch_bounds__(256) void add_ln_kernel(const float* x, const float* r,
                                                     const float* w, const float* b,
                                                     float* out) {
  const long t = blockIdx.x;
  const float* px = x + t * DMODEL;
  const float* pr = r + t * DMODEL;
  float v[4]; float s = 0.f;
  #pragma unroll
  for (int i = 0; i < 4; ++i) {
    int d = threadIdx.x + i * 256;
    v[i] = px[d] + pr[d];
    s += v[i];
  }
  float mean = blockSum(s) * (1.0f / DMODEL);
  float ss = 0.f;
  #pragma unroll
  for (int i = 0; i < 4; ++i) { float dv = v[i] - mean; ss += dv * dv; }
  float var = blockSum(ss) * (1.0f / DMODEL);
  float rs = rsqrtf(var + 1e-5f);
  #pragma unroll
  for (int i = 0; i < 4; ++i) {
    int d = threadIdx.x + i * 256;
    out[t * DMODEL + d] = (v[i] - mean) * rs * w[d] + b[d];
  }
}

// ---------- router: logits, softmax, top-2, stats ----------
__global__ __launch_bounds__(256) void router_kernel(const float* x1, const float* rw,
                                                     int* idx0, int* idx1,
                                                     float* g0, float* g1,
                                                     float* probSum, float* lseSq) {
  const int lane = threadIdx.x & 31;
  const long t = (long)blockIdx.x * 8 + (threadIdx.x >> 5);
  float acc[8] = {0.f,0.f,0.f,0.f,0.f,0.f,0.f,0.f};
  for (int d = lane; d < DMODEL; d += 32) {
    float xv = x1[t * DMODEL + d];
    const f32x4* r4 = (const f32x4*)(rw + d * NEXP);
    f32x4 ra = r4[0], rb = r4[1];
    acc[0] += xv * ra.x; acc[1] += xv * ra.y; acc[2] += xv * ra.z; acc[3] += xv * ra.w;
    acc[4] += xv * rb.x; acc[5] += xv * rb.y; acc[6] += xv * rb.z; acc[7] += xv * rb.w;
  }
  #pragma unroll
  for (int e = 0; e < 8; ++e) {
    #pragma unroll
    for (int o = 16; o > 0; o >>= 1) acc[e] += __shfl_down(acc[e], o);
  }
  if (lane == 0) {
    float mx = acc[0];
    #pragma unroll
    for (int e = 1; e < 8; ++e) mx = fmaxf(mx, acc[e]);
    float p[8], se = 0.f;
    #pragma unroll
    for (int e = 0; e < 8; ++e) { p[e] = expf(acc[e] - mx); se += p[e]; }
    float inv = 1.0f / se;
    #pragma unroll
    for (int e = 0; e < 8; ++e) p[e] *= inv;
    float lse = mx + logf(se);
    int i0 = 0;
    #pragma unroll
    for (int e = 1; e < 8; ++e) if (p[e] > p[i0]) i0 = e;
    int i1 = (i0 == 0) ? 1 : 0;
    #pragma unroll
    for (int e = 0; e < 8; ++e) if (e != i0 && p[e] > p[i1]) i1 = e;
    idx0[t] = i0; idx1[t] = i1; g0[t] = p[i0]; g1[t] = p[i1];
    #pragma unroll
    for (int e = 0; e < 8; ++e) atomicAdd(&probSum[e], p[e]);
    atomicAdd(lseSq, lse * lse);
  }
}

// ---------- per-expert capacity scan (matches cumsum semantics) ----------
__global__ void scan_kernel(const int* idx0, const int* idx1,
                            int* slot0, int* slot1, int* count0) {
  const int e = threadIdx.x;
  if (e >= NEXP) return;
  int c0 = 0;
  for (int t = 0; t < TOK; ++t)
    if (idx0[t] == e) { slot0[t] = (c0 < CAP) ? c0 : -1; c0++; }
  count0[e] = c0;
  int c1 = 0;
  for (int t = 0; t < TOK; ++t)
    if (idx1[t] == e) { int pos = c1 + c0; slot1[t] = (pos < CAP) ? pos : -1; c1++; }
}

// ---------- scatter tokens into expert slots ----------
__global__ __launch_bounds__(256) void dispatch_kernel(const float* x1,
                                                       const int* idx0, const int* idx1,
                                                       const int* slot0, const int* slot1,
                                                       float* xin) {
  const long t = blockIdx.x;
  const int j = blockIdx.y;
  const int e = j ? idx1[t] : idx0[t];
  const int s = j ? slot1[t] : slot0[t];
  if (s < 0) return;
  const f32x4* src = (const f32x4*)(x1 + t * DMODEL);
  f32x4* dst = (f32x4*)(xin + ((long)e * CAP + s) * DMODEL);
  dst[threadIdx.x] = src[threadIdx.x];
}

// ---------- combine expert outputs + residual + LN2 -> d_out ----------
__global__ __launch_bounds__(256) void combine_ln_kernel(const float* x1, const float* out_e,
                                                         const int* idx0, const int* idx1,
                                                         const int* slot0, const int* slot1,
                                                         const float* g0, const float* g1,
                                                         const float* w, const float* b,
                                                         float* out) {
  const long t = blockIdx.x;
  const int e0 = idx0[t], e1 = idx1[t], s0 = slot0[t], s1 = slot1[t];
  const float gg0 = g0[t], gg1 = g1[t];
  const float* p0 = (s0 >= 0) ? out_e + ((long)e0 * CAP + s0) * DMODEL : nullptr;
  const float* p1 = (s1 >= 0) ? out_e + ((long)e1 * CAP + s1) * DMODEL : nullptr;
  float v[4]; float s = 0.f;
  #pragma unroll
  for (int i = 0; i < 4; ++i) {
    int d = threadIdx.x + i * 256;
    float y = x1[t * DMODEL + d];
    if (p0) y += gg0 * p0[d];
    if (p1) y += gg1 * p1[d];
    v[i] = y; s += y;
  }
  float mean = blockSum(s) * (1.0f / DMODEL);
  float ss = 0.f;
  #pragma unroll
  for (int i = 0; i < 4; ++i) { float dv = v[i] - mean; ss += dv * dv; }
  float var = blockSum(ss) * (1.0f / DMODEL);
  float rs = rsqrtf(var + 1e-5f);
  #pragma unroll
  for (int i = 0; i < 4; ++i) {
    int d = threadIdx.x + i * 256;
    out[t * DMODEL + d] = (v[i] - mean) * rs * w[d] + b[d];
  }
}

// ---------- aux loss ----------
__global__ void aux_kernel(const int* count0, const float* probSum, const float* lseSq,
                           float* out) {
  float bal = 0.f;
  for (int e = 0; e < NEXP; ++e)
    bal += ((float)count0[e] * (1.0f / TOK)) * (probSum[e] * (1.0f / TOK));
  bal *= (float)NEXP;
  float z = lseSq[0] * (1.0f / TOK);
  out[(long)TOK * DMODEL] = 0.01f * bal + 0.001f * z;
}

__global__ void zero_kernel(float* p, long n) {
  long i = (long)blockIdx.x * 256 + threadIdx.x;
  if (i < n) p[i] = 0.f;
}

// ---------- host-side orchestration ----------
static void launch_gemm(hipStream_t st, const float* A, const float* B, const float* bias,
                        float* C, int M, int N, int K, int lda, int bsk, int bsn, int ldc,
                        int Z, int zdiv, long aO, long aI, long bO, long bI,
                        long cO, long cI, long biasO, long biasI, int epi) {
  GemmP p;
  p.A = A; p.B = B; p.bias = bias; p.C = C;
  p.K = K; p.lda = lda; p.bsk = bsk; p.bsn = bsn; p.ldc = ldc;
  p.zdiv = zdiv; p.epi = epi; p.pad = 0;
  p.aO = aO; p.aI = aI; p.bO = bO; p.bI = bI;
  p.cO = cO; p.cI = cI; p.biasO = biasO; p.biasI = biasI;
  dim3 g(N / 64, M / 128, Z), blk(256);
  if (bsn == 1) gemm_bf16_wmma<true ><<<g, blk, 0, st>>>(p);
  else          gemm_bf16_wmma<false><<<g, blk, 0, st>>>(p);
}

extern "C" void kernel_launch(void* const* d_in, const int* in_sizes, int n_in,
                              void* d_out, int out_size, void* d_ws, size_t ws_size,
                              hipStream_t stream) {
  (void)in_sizes; (void)n_in; (void)out_size; (void)ws_size;

  const float* q_in   = (const float*)d_in[0];
  const float* k_in   = (const float*)d_in[1];
  const float* v_in   = (const float*)d_in[2];
  const float* in_w   = (const float*)d_in[3];
  const float* in_b   = (const float*)d_in[4];
  const float* out_w  = (const float*)d_in[5];
  const float* out_b  = (const float*)d_in[6];
  const float* ln1_w  = (const float*)d_in[7];
  const float* ln1_b  = (const float*)d_in[8];
  const float* ln2_w  = (const float*)d_in[9];
  const float* ln2_b  = (const float*)d_in[10];
  const float* rw     = (const float*)d_in[11];
  const float* w1     = (const float*)d_in[12];
  const float* b1     = (const float*)d_in[13];
  const float* w2     = (const float*)d_in[14];
  const float* b2     = (const float*)d_in[15];
  float* out = (float*)d_out;
  float* ws  = (float*)d_ws;

  // workspace layout (float offsets)
  const long TD = (long)TOK * DMODEL;             // 4,194,304
  long o_qp   = 0;
  long o_kp   = o_qp  + TD;
  long o_vp   = o_kp  + TD;
  long o_att  = o_vp  + TD;
  long o_attp = o_att + TD;
  long o_x1   = o_attp+ TD;
  long o_xin  = o_x1  + TD;
  long o_oute = o_xin + (long)NEXP * CAP * DMODEL;
  long o_h    = o_oute+ (long)NEXP * CAP * DMODEL;
  long o_sc   = o_h   + (long)NEXP * CAP * DFF;
  long o_misc = o_sc  + (long)NHEAD * 8 * SEQ * SEQ;   // 128 bh * 512 * 512

  int*   idx0    = (int*)  (ws + o_misc);
  int*   idx1    = idx0 + TOK;
  int*   slot0   = idx1 + TOK;
  int*   slot1   = slot0 + TOK;
  int*   count0  = slot1 + TOK;
  float* g0      = (float*)(count0 + NEXP);
  float* g1      = g0 + TOK;
  float* probSum = g1 + TOK;       // 8 floats
  float* lseSq   = probSum + NEXP; // 1 float

  // 1) QKV projections: x @ W^T + b   (M=4096,N=1024,K=1024)
  const float* ins[3] = { q_in, k_in, v_in };
  float* prj[3] = { ws + o_qp, ws + o_kp, ws + o_vp };
  for (int i = 0; i < 3; ++i) {
    launch_gemm(stream, ins[i], in_w + (long)i * DMODEL * DMODEL, in_b + i * DMODEL,
                prj[i], TOK, DMODEL, DMODEL,
                /*lda*/DMODEL, /*bsk*/1, /*bsn*/DMODEL, /*ldc*/DMODEL,
                /*Z*/1, /*zdiv*/1, 0,0, 0,0, 0,0, 0,0, /*epi*/0);
  }

  // 2) scores = qh @ kh^T, batched over (b,h) via grid.z  (M=N=512, K=64)
  launch_gemm(stream, ws + o_qp, ws + o_kp, nullptr, ws + o_sc,
              SEQ, SEQ, DHEAD,
              DMODEL, /*bsk*/1, /*bsn*/DMODEL, /*ldc*/SEQ,
              /*Z*/8 * NHEAD, /*zdiv*/NHEAD,
              (long)SEQ * DMODEL, DHEAD,
              (long)SEQ * DMODEL, DHEAD,
              (long)NHEAD * SEQ * SEQ, (long)SEQ * SEQ, 0, 0, 0);

  // 3) softmax with 1/8 scale, in place (65536 rows)
  softmax512_kernel<<<8 * NHEAD * SEQ, 256, 0, stream>>>(ws + o_sc);

  // 4) o = attn @ vh  (M=512,N=64,K=512), write into head-interleaved layout
  launch_gemm(stream, ws + o_sc, ws + o_vp, nullptr, ws + o_att,
              SEQ, DHEAD, SEQ,
              /*lda*/SEQ, /*bsk*/DMODEL, /*bsn*/1, /*ldc*/DMODEL,
              /*Z*/8 * NHEAD, /*zdiv*/NHEAD,
              (long)NHEAD * SEQ * SEQ, (long)SEQ * SEQ,
              (long)SEQ * DMODEL, DHEAD,
              (long)SEQ * DMODEL, DHEAD, 0, 0, 0);

  // 5) out-projection
  launch_gemm(stream, ws + o_att, out_w, out_b, ws + o_attp,
              TOK, DMODEL, DMODEL,
              DMODEL, 1, DMODEL, DMODEL,
              1, 1, 0,0, 0,0, 0,0, 0,0, 0);

  // 6) x1 = LN(q + attn)
  add_ln_kernel<<<TOK, 256, 0, stream>>>(q_in, ws + o_attp, ln1_w, ln1_b, ws + o_x1);

  // 7) zero MoE scratch (xin + stats accumulators)
  {
    long nz = (long)NEXP * CAP * DMODEL;
    zero_kernel<<<(unsigned)((nz + 255) / 256), 256, 0, stream>>>(ws + o_xin, nz);
    zero_kernel<<<1, 256, 0, stream>>>(probSum, NEXP + 1);
  }

  // 8) router + 9) capacity scan + 10) dispatch
  router_kernel<<<TOK / 8, 256, 0, stream>>>(ws + o_x1, rw, idx0, idx1, g0, g1,
                                             probSum, lseSq);
  scan_kernel<<<1, 32, 0, stream>>>(idx0, idx1, slot0, slot1, count0);
  dispatch_kernel<<<dim3(TOK, 2), 256, 0, stream>>>(ws + o_x1, idx0, idx1, slot0, slot1,
                                                    ws + o_xin);

  // 11) expert FF1: gelu(xin @ w1 + b1)   (per expert: M=1280,N=4096,K=1024)
  launch_gemm(stream, ws + o_xin, w1, b1, ws + o_h,
              CAP, DFF, DMODEL,
              DMODEL, /*bsk*/DFF, /*bsn*/1, /*ldc*/DFF,
              NEXP, 1,
              (long)CAP * DMODEL, 0,
              (long)DMODEL * DFF, 0,
              (long)CAP * DFF, 0,
              DFF, 0, /*epi gelu*/1);

  // 12) expert FF2: h @ w2 + b2          (per expert: M=1280,N=1024,K=4096)
  launch_gemm(stream, ws + o_h, w2, b2, ws + o_oute,
              CAP, DMODEL, DFF,
              DFF, /*bsk*/DMODEL, /*bsn*/1, /*ldc*/DMODEL,
              NEXP, 1,
              (long)CAP * DFF, 0,
              (long)DFF * DMODEL, 0,
              (long)CAP * DMODEL, 0,
              DMODEL, 0, 0);

  // 13) combine + residual + LN2 -> d_out
  combine_ln_kernel<<<TOK, 256, 0, stream>>>(ws + o_x1, ws + o_oute, idx0, idx1,
                                             slot0, slot1, g0, g1, ln2_w, ln2_b, out);

  // 14) aux loss scalar -> d_out[TOK*DMODEL]
  aux_kernel<<<1, 1, 0, stream>>>(count0, probSum, lseSq, out);
}
